// TransolverBlock_55207509623217
// MI455X (gfx1250) — compile-verified, matrix operand order checked by reading, not run on hardware
//
#include <hip/hip_runtime.h>
#include <hip/hip_bf16.h>

typedef unsigned short u16;
typedef unsigned int   u32;

typedef __attribute__((ext_vector_type(16))) __bf16 v16bf;
typedef __attribute__((ext_vector_type(8)))  float  v8f;

#define LDB 264   // bf16 LDS row stride (ushorts), 16B-aligned rows, bank-rotating
#define LDF 260   // f32  LDS row stride (floats),  16B-aligned rows

// ---------- workspace layout (bytes) ----------
#define WS_WFX    ((size_t)0)         // 256x256 bf16 swizzled  (131072)
#define WS_WX     ((size_t)131072)
#define WS_WOUT   ((size_t)262144)
#define WS_WPRE   ((size_t)393216)    // 256x1024 bf16          (524288)
#define WS_WPOST  ((size_t)917504)    // 1024x256 bf16          (524288)
#define WS_WSL    ((size_t)1441792)   // 32x32 bf16             (2048)
#define WS_STOK   ((size_t)1443840)   // B*H*G*D f32 accum      (65536)
#define WS_SNORM  ((size_t)1509376)   // B*H*G f32 accum        (2048)
#define WS_OSL    ((size_t)1511424)   // out_slice swizzled bf16(32768)
#define WS_SW     ((size_t)1572864)   // sw bf16 [B*N][H*G]     (33554432)

__device__ __forceinline__ u16 f2bf(float f) {
  u32 u = __float_as_uint(f);
  u32 r = u + 0x7FFFu + ((u >> 16) & 1u);
  return (u16)(r >> 16);
}
__device__ __forceinline__ float bf2f(u16 h) { return __uint_as_float(((u32)h) << 16); }

union FragU { uint4 q[2]; v16bf v; };

__device__ __forceinline__ v8f vzero() { v8f z; for (int i = 0; i < 8; ++i) z[i] = 0.f; return z; }

// A fragment (16x32 bf16) from row-major LDS. ISA layout: lanes0-15 M=0..15 hold
// K = kb+0..7 then kb+16..23 ; lanes16-31 hold K = kb+8..15 then kb+24..31.
__device__ __forceinline__ v16bf frag_A_lds(const u16* base, int row0, int ld, int kbase) {
  int lane = threadIdx.x & 31;
  const u16* p = base + (row0 + (lane & 15)) * ld + kbase + ((lane & 16) ? 8 : 0);
  FragU f;
  f.q[0] = *(const uint4*)(p);
  f.q[1] = *(const uint4*)(p + 16);
  return f.v;
}

// B fragment from pre-swizzled global weights: frag = 512 bf16, lane-major (32B/lane).
__device__ __forceinline__ v16bf frag_B_glb(const u16* wsw, int fragIdx) {
  const u16* p = wsw + ((size_t)fragIdx << 9) + ((threadIdx.x & 31) << 4);
  FragU f;
  f.q[0] = *(const uint4*)(p);
  f.q[1] = *(const uint4*)(p + 8);
  return f.v;
}

__device__ __forceinline__ v8f wmma_bf16(v16bf a, v16bf b, v8f c) {
  return __builtin_amdgcn_wmma_f32_16x16x32_bf16(false, a, false, b, (short)0, c, false, false);
}

// Software-pipelined K=256 GEMM tile: acc[j] += A(rows rt*16.., K=ks*32..) x
// W(frag fragBase + ks*fragStrideKs + ch*8 + j). B fragments are streamed with a
// 2-deep rotating register buffer (issued two WMMAs ahead) and the next k-step's
// A fragment is loaded from LDS before the WMMA burst, so the per-WMMA
// s_wait_loadcnt never has to drain to zero.
__device__ __forceinline__ void gemm_acc8(v8f acc[8], const u16* A, int rt,
                                          const u16* W, int fragBase, int fragStrideKs,
                                          int ch) {
  v16bf a  = frag_A_lds(A, rt * 16, LDB, 0);
  v16bf an = a;
  v16bf b0 = frag_B_glb(W, fragBase + ch * 8 + 0);
  v16bf b1 = frag_B_glb(W, fragBase + ch * 8 + 1);
#pragma unroll
  for (int ks = 0; ks < 8; ++ks) {
    if (ks < 7)
      an = frag_A_lds(A, rt * 16, LDB, (ks + 1) * 32);
    __builtin_prefetch((const void*)(W + ((size_t)(fragBase + ((ks + 1) & 7) * fragStrideKs + ch * 8) << 9)), 0, 1);
#pragma unroll
    for (int j = 0; j < 8; ++j) {
      v16bf bn = b1;
      int t2 = ks * 8 + j + 2;                 // stream position two ahead
      if (t2 < 64) {
        int ks2 = t2 >> 3, j2 = t2 & 7;
        b1 = frag_B_glb(W, fragBase + ks2 * fragStrideKs + ch * 8 + j2);
      }
      acc[j] = wmma_bf16(a, b0, acc[j]);
      b0 = bn;
    }
    a = an;
  }
}

// LayerNorm over 256 cols for a 64-token tile; 4 threads/token.
__device__ __forceinline__ void ln_block(const float* src, int sld,
                                         const float* gs, const float* gb,
                                         u16* dst, int dld,
                                         float (*red)[4][2], float (*mstat)[2]) {
  int tid = threadIdx.x, tok = tid >> 2, qtr = tid & 3;
  const float* prow = src + tok * sld + qtr * 64;
  float s = 0.f, sq = 0.f;
  for (int i = 0; i < 16; ++i) {
    float4 v = *(const float4*)(prow + i * 4);
    s  += v.x + v.y + v.z + v.w;
    sq += v.x * v.x + v.y * v.y + v.z * v.z + v.w * v.w;
  }
  red[tok][qtr][0] = s; red[tok][qtr][1] = sq;
  __syncthreads();
  if (qtr == 0) {
    float ts = 0.f, tq = 0.f;
    for (int q = 0; q < 4; ++q) { ts += red[tok][q][0]; tq += red[tok][q][1]; }
    float m = ts * (1.f / 256.f);
    float va = tq * (1.f / 256.f) - m * m;
    mstat[tok][0] = m; mstat[tok][1] = rsqrtf(va + 1e-5f);
  }
  __syncthreads();
  float m = mstat[tok][0], rs = mstat[tok][1];
  for (int i = 0; i < 16; ++i) {
    float4 v = *(const float4*)(prow + i * 4);
    int c = qtr * 64 + i * 4;
    dst[tok * dld + c + 0] = f2bf((v.x - m) * rs * gs[c + 0] + gb[c + 0]);
    dst[tok * dld + c + 1] = f2bf((v.y - m) * rs * gs[c + 1] + gb[c + 1]);
    dst[tok * dld + c + 2] = f2bf((v.z - m) * rs * gs[c + 2] + gb[c + 2]);
    dst[tok * dld + c + 3] = f2bf((v.w - m) * rs * gs[c + 3] + gb[c + 3]);
  }
}

// ---------------- stage 0: weight repack + accumulator zero ----------------
__global__ __launch_bounds__(256) void k0_repack(const float* __restrict__ src,
                                                 u16* __restrict__ dst, int K, int Nmat) {
  int idx = blockIdx.x * 256 + threadIdx.x;
  if (idx >= K * Nmat) return;
  int frag = idx >> 9, rem = idx & 511;
  int lane = rem >> 4, e = rem & 15;
  int NT = Nmat >> 4;
  int kt = frag / NT, nt = frag % NT;
  int Ksrc = kt * 32 + ((lane & 16) ? 8 : 0) + ((e >= 8) ? 16 : 0) + (e & 7);
  int Nsrc = nt * 16 + (lane & 15);
  dst[idx] = f2bf(src[Ksrc * Nmat + Nsrc]);
}

__global__ __launch_bounds__(256) void k0_zero(float* __restrict__ p, int n) {
  int i = blockIdx.x * 256 + threadIdx.x;
  if (i < n) p[i] = 0.f;
}

// ---------------- stage 1: LN1 + projections + slice softmax + pooled accum ----------------
__global__ __launch_bounds__(256) void k1_proj(
    const float* __restrict__ fx, const float* __restrict__ ln1_s, const float* __restrict__ ln1_b,
    const float* __restrict__ bfx, const float* __restrict__ bx,
    const float* __restrict__ bsl, const float* __restrict__ temp,
    const u16* __restrict__ wfx_sw, const u16* __restrict__ wx_sw, const u16* __restrict__ wsl_sw,
    u16* __restrict__ sw_g, float* __restrict__ stok_acc, float* __restrict__ snorm_acc)
{
  __shared__ __align__(16) float R0f[16640];      // h1 bf16, then logits f32
  __shared__ __align__(16) u16   R1[64 * LDB];    // fx_mid bf16
  __shared__ __align__(16) u16   R2[64 * LDB];    // x_mid bf16, then sw bf16
  __shared__ float red[64][4][2];
  __shared__ float mstat[64][2];

  const int tid = threadIdx.x;
  const int wave = tid >> 5, lane = tid & 31;
  const int rt = wave & 3, ch = wave >> 2;
  const int hh = (lane & 16) ? 8 : 0;

  const int tileId = blockIdx.x;
  const int b = tileId >> 9;
  const int n0 = (tileId & 511) << 6;
  const float* fxp = fx + ((size_t)(b * 32768 + n0)) * 256;

  u16* h1 = (u16*)R0f;
  float* Dl = R0f;

  // LN1 -> h1 (bf16)
  ln_block(fxp, 256, ln1_s, ln1_b, h1, LDB, red, mstat);
  __syncthreads();

  // fx_mid = h1 @ Wfx + bfx  -> R1 (bf16)
  {
    v8f acc[8]; for (int j = 0; j < 8; ++j) acc[j] = vzero();
    gemm_acc8(acc, h1, rt, wfx_sw, 0, 16, ch);
    for (int j = 0; j < 8; ++j) {
      int col = (ch * 8 + j) * 16 + (lane & 15);
      float bv = bfx[col];
      for (int r = 0; r < 8; ++r)
        R1[(rt * 16 + hh + r) * LDB + col] = f2bf(acc[j][r] + bv);
    }
  }
  // x_mid = h1 @ Wx + bx  -> R2 (bf16)
  {
    v8f acc[8]; for (int j = 0; j < 8; ++j) acc[j] = vzero();
    gemm_acc8(acc, h1, rt, wx_sw, 0, 16, ch);
    for (int j = 0; j < 8; ++j) {
      int col = (ch * 8 + j) * 16 + (lane & 15);
      float bv = bx[col];
      for (int r = 0; r < 8; ++r)
        R2[(rt * 16 + hh + r) * LDB + col] = f2bf(acc[j][r] + bv);
    }
  }
  __syncthreads();   // h1 dead; R2 complete

  // slice logits: per head, x_mid[:, h*32..] @ Wsl  -> Dl (f32, col = h*32+g)
  for (int j = 0; j < 8; ++j) {
    int ct = ch * 8 + j;
    int h = ct >> 1, nt = ct & 1;
    v16bf a = frag_A_lds(R2, rt * 16, LDB, h * 32);
    v16bf bb = frag_B_glb(wsl_sw, nt);
    v8f z = vzero();
    z = wmma_bf16(a, bb, z);
    int g = nt * 16 + (lane & 15);
    float bv = bsl[g];
    for (int r = 0; r < 8; ++r)
      Dl[(rt * 16 + hh + r) * LDF + ct * 16 + (lane & 15)] = z[r] + bv;
  }
  __syncthreads();

  // softmax over G with temperature; write sw bf16 to R2 (x_mid dead) + global
  for (int i = 0; i < 2; ++i) {
    int rr = tid + i * 256;          // 0..511 = (token, head)
    int n = rr >> 3, h = rr & 7;
    float invt = 1.0f / temp[h];
    const float* lrow = Dl + n * LDF + h * 32;
    float lv[32];
    float mx = -1e30f;
    for (int g = 0; g < 32; ++g) { lv[g] = lrow[g] * invt; mx = fmaxf(mx, lv[g]); }
    float ssum = 0.f;
    for (int g = 0; g < 32; ++g) { lv[g] = __expf(lv[g] - mx); ssum += lv[g]; }
    float inv = 1.f / ssum;
    u16* srow = R2 + n * LDB + h * 32;
    u16* grow = sw_g + ((size_t)(b * 32768 + n0 + n)) * 256 + h * 32;
    for (int g = 0; g < 32; ++g) {
      u16 hv = f2bf(lv[g] * inv);
      srow[g] = hv; grow[g] = hv;
    }
  }
  __syncthreads();

  // pooled slice tokens: thread = (h,g); lanes of a wave share h -> LDS broadcast on fx_mid
  {
    int h = tid >> 5, g = tid & 31;
    float accd[32];
    for (int d = 0; d < 32; ++d) accd[d] = 0.f;
    float sn = 0.f;
    for (int n = 0; n < 64; ++n) {
      float s = bf2f(R2[n * LDB + h * 32 + g]);
      sn += s;
      const uint4* pr = (const uint4*)(R1 + n * LDB + h * 32);
      for (int c = 0; c < 4; ++c) {
        uint4 q = pr[c];
        u32 w0 = q.x, w1 = q.y, w2 = q.z, w3 = q.w;
        accd[c*8+0] += s * bf2f((u16)(w0 & 0xFFFF)); accd[c*8+1] += s * bf2f((u16)(w0 >> 16));
        accd[c*8+2] += s * bf2f((u16)(w1 & 0xFFFF)); accd[c*8+3] += s * bf2f((u16)(w1 >> 16));
        accd[c*8+4] += s * bf2f((u16)(w2 & 0xFFFF)); accd[c*8+5] += s * bf2f((u16)(w2 >> 16));
        accd[c*8+6] += s * bf2f((u16)(w3 & 0xFFFF)); accd[c*8+7] += s * bf2f((u16)(w3 >> 16));
      }
    }
    float* sa = stok_acc + (((b * 8 + h) * 32 + g) << 5);
    for (int d = 0; d < 32; ++d) atomicAdd(sa + d, accd[d]);
    atomicAdd(snorm_acc + (b * 8 + h) * 32 + g, sn);
  }
}

// ---------------- stage 2: tiny 32x32 attention over slice tokens ----------------
__global__ __launch_bounds__(256) void k2_attn(
    const float* __restrict__ stok_acc, const float* __restrict__ snorm_acc,
    const float* __restrict__ Wq, const float* __restrict__ Wk, const float* __restrict__ Wv,
    u16* __restrict__ osl_sw)
{
  __shared__ float st[32][32], qm[32][32], km[32][32], vm[32][32], sc[32][32], os[32][32];
  const int bh = blockIdx.x;           // (b*8 + h)
  const int tid = threadIdx.x;
  const float* sa = stok_acc + bh * 1024;
  const float* sn = snorm_acc + bh * 32;

  for (int i = 0; i < 4; ++i) {
    int idx = tid + i * 256, g = idx >> 5, d = idx & 31;
    st[g][d] = sa[idx] / (sn[g] + 1e-5f);
  }
  __syncthreads();
  for (int i = 0; i < 4; ++i) {
    int idx = tid + i * 256, g = idx >> 5, d = idx & 31;
    float aq = 0.f, ak = 0.f, av = 0.f;
    for (int e = 0; e < 32; ++e) {
      float s = st[g][e];
      aq += s * Wq[e * 32 + d]; ak += s * Wk[e * 32 + d]; av += s * Wv[e * 32 + d];
    }
    qm[g][d] = aq; km[g][d] = ak; vm[g][d] = av;
  }
  __syncthreads();
  for (int i = 0; i < 4; ++i) {
    int idx = tid + i * 256, g = idx >> 5, j = idx & 31;
    float a = 0.f;
    for (int e = 0; e < 32; ++e) a += qm[g][e] * km[j][e];
    sc[g][j] = a * 0.17677669529663687f;   // 1/sqrt(32)
  }
  __syncthreads();
  if (tid < 32) {
    float mx = -1e30f;
    for (int j = 0; j < 32; ++j) mx = fmaxf(mx, sc[tid][j]);
    float s = 0.f;
    for (int j = 0; j < 32; ++j) { float e = __expf(sc[tid][j] - mx); sc[tid][j] = e; s += e; }
    float inv = 1.f / s;
    for (int j = 0; j < 32; ++j) sc[tid][j] *= inv;
  }
  __syncthreads();
  for (int i = 0; i < 4; ++i) {
    int idx = tid + i * 256, g = idx >> 5, d = idx & 31;
    float a = 0.f;
    for (int j = 0; j < 32; ++j) a += sc[g][j] * vm[j][d];
    os[g][d] = a;
  }
  __syncthreads();
  // emit out_slice as swizzled bf16 B-fragments (K=g, N=d), 2 frags per (b,h)
  if (tid < 64) {
    int nt = tid >> 5, lane = tid & 31;
    u16* dst = osl_sw + ((size_t)(bh * 2 + nt) << 9) + lane * 16;
    for (int e = 0; e < 16; ++e) {
      int K = ((lane & 16) ? 8 : 0) + ((e >= 8) ? 16 : 0) + (e & 7);
      dst[e] = f2bf(os[K][nt * 16 + (lane & 15)]);
    }
  }
}

// ---------------- stage 3: broadcast + out-proj + residual + LN2 + MLP ----------------
__global__ __launch_bounds__(256) void k3_out(
    const float* __restrict__ fx,
    const u16* __restrict__ sw_g, const u16* __restrict__ osl_sw,
    const u16* __restrict__ wout_sw, const float* __restrict__ bout,
    const float* __restrict__ ln2_s, const float* __restrict__ ln2_b,
    const u16* __restrict__ wpre_sw, const float* __restrict__ bpre,
    const u16* __restrict__ wpost_sw, const float* __restrict__ bpost,
    float* __restrict__ out)
{
  __shared__ __align__(16) u16   Bsw[64 * LDB];   // sw tile; later GELU chunk
  __shared__ __align__(16) u16   Bx[64 * LDB];    // out_x bf16; later h bf16
  __shared__ __align__(16) float Bfx2[64 * LDF];  // residual stream f32
  __shared__ float red[64][4][2];
  __shared__ float mstat[64][2];

  const int tid = threadIdx.x;
  const int wave = tid >> 5, lane = tid & 31;
  const int rt = wave & 3, ch = wave >> 2;
  const int hh = (lane & 16) ? 8 : 0;

  const int tileId = blockIdx.x;
  const int b = tileId >> 9;
  const int n0 = (tileId & 511) << 6;

  // A: stage sw tile into LDS
  {
    const uint4* src = (const uint4*)(sw_g + ((size_t)(b * 32768 + n0)) * 256);
    for (int i = tid; i < 64 * 32; i += 256) {
      int row = i >> 5, c = i & 31;
      *(uint4*)(Bsw + row * LDB + c * 8) = src[row * 32 + c];
    }
  }
  __syncthreads();

  // B: out_x = sw @ out_slice (per head, K=32)  -> Bx bf16
  for (int j = 0; j < 8; ++j) {
    int ct = ch * 8 + j;
    int h = ct >> 1, nt = ct & 1;
    v16bf a = frag_A_lds(Bsw, rt * 16, LDB, h * 32);
    v16bf bb = frag_B_glb(osl_sw, (b * 8 + h) * 2 + nt);
    v8f z = vzero();
    z = wmma_bf16(a, bb, z);
    int col = ct * 16 + (lane & 15);
    for (int r = 0; r < 8; ++r)
      Bx[(rt * 16 + hh + r) * LDB + col] = f2bf(z[r]);
  }
  __syncthreads();

  // C: fx2 = out_x @ Wout + bout + fx  -> Bfx2 f32
  {
    v8f acc[8]; for (int j = 0; j < 8; ++j) acc[j] = vzero();
    gemm_acc8(acc, Bx, rt, wout_sw, 0, 16, ch);
    for (int j = 0; j < 8; ++j) {
      int col = (ch * 8 + j) * 16 + (lane & 15);
      float bv = bout[col];
      for (int r = 0; r < 8; ++r) {
        int row = rt * 16 + hh + r;
        float v = acc[j][r] + bv + fx[((size_t)(b * 32768 + n0 + row)) * 256 + col];
        Bfx2[row * LDF + col] = v;
      }
    }
  }
  __syncthreads();

  // D: LN2 -> h bf16 into Bx (out_x dead)
  ln_block(Bfx2, LDF, ln2_s, ln2_b, Bx, LDB, red, mstat);
  __syncthreads();

  // E: MLP 256 -> 1024 -> 256, chunked 4x256 over hidden dim; final acc in registers
  v8f facc[8]; for (int j = 0; j < 8; ++j) facc[j] = vzero();
  for (int c = 0; c < 4; ++c) {
    v8f acc[8]; for (int j = 0; j < 8; ++j) acc[j] = vzero();
    gemm_acc8(acc, Bx, rt, wpre_sw, c * 16, 64, ch);
    for (int j = 0; j < 8; ++j) {
      int colL = (ch * 8 + j) * 16 + (lane & 15);
      float bv = bpre[c * 256 + colL];
      for (int r = 0; r < 8; ++r) {
        float xv = acc[j][r] + bv;
        float gl = 0.5f * xv * (1.0f + erff(xv * 0.70710678118654752f));
        Bsw[(rt * 16 + hh + r) * LDB + colL] = f2bf(gl);
      }
    }
    __syncthreads();
    gemm_acc8(facc, Bsw, rt, wpost_sw, c * 128, 16, ch);
    __syncthreads();
  }

  // F: out = mlp + bpost + fx2
  for (int j = 0; j < 8; ++j) {
    int col = (ch * 8 + j) * 16 + (lane & 15);
    float bv = bpost[col];
    for (int r = 0; r < 8; ++r) {
      int row = rt * 16 + hh + r;
      out[((size_t)(b * 32768 + n0 + row)) * 256 + col] = facc[j][r] + bv + Bfx2[row * LDF + col];
    }
  }
}

extern "C" void kernel_launch(void* const* d_in, const int* in_sizes, int n_in,
                              void* d_out, int out_size, void* d_ws, size_t ws_size,
                              hipStream_t stream) {
  const float* fx    = (const float*)d_in[0];
  const float* ln1_s = (const float*)d_in[1];
  const float* ln1_b = (const float*)d_in[2];
  const float* Wfx   = (const float*)d_in[3];
  const float* bfx   = (const float*)d_in[4];
  const float* Wx    = (const float*)d_in[5];
  const float* bx    = (const float*)d_in[6];
  const float* Wsl   = (const float*)d_in[7];
  const float* bsl   = (const float*)d_in[8];
  const float* temp  = (const float*)d_in[9];
  const float* Wq    = (const float*)d_in[10];
  const float* Wk    = (const float*)d_in[11];
  const float* Wv    = (const float*)d_in[12];
  const float* Wout  = (const float*)d_in[13];
  const float* bout  = (const float*)d_in[14];
  const float* ln2_s = (const float*)d_in[15];
  const float* ln2_b = (const float*)d_in[16];
  const float* Wpre  = (const float*)d_in[17];
  const float* bpre  = (const float*)d_in[18];
  const float* Wpost = (const float*)d_in[19];
  const float* bpost = (const float*)d_in[20];

  char* ws = (char*)d_ws;
  u16*   wfx_sw   = (u16*)(ws + WS_WFX);
  u16*   wx_sw    = (u16*)(ws + WS_WX);
  u16*   wout_sw  = (u16*)(ws + WS_WOUT);
  u16*   wpre_sw  = (u16*)(ws + WS_WPRE);
  u16*   wpost_sw = (u16*)(ws + WS_WPOST);
  u16*   wsl_sw   = (u16*)(ws + WS_WSL);
  float* stok_acc = (float*)(ws + WS_STOK);
  float* snorm    = (float*)(ws + WS_SNORM);
  u16*   osl_sw   = (u16*)(ws + WS_OSL);
  u16*   sw_g     = (u16*)(ws + WS_SW);

  // stage 0: repack weights to bf16 swizzled fragments + zero accumulators
  k0_repack<<<256,  256, 0, stream>>>(Wfx,   wfx_sw,   256, 256);
  k0_repack<<<256,  256, 0, stream>>>(Wx,    wx_sw,    256, 256);
  k0_repack<<<256,  256, 0, stream>>>(Wout,  wout_sw,  256, 256);
  k0_repack<<<1024, 256, 0, stream>>>(Wpre,  wpre_sw,  256, 1024);
  k0_repack<<<1024, 256, 0, stream>>>(Wpost, wpost_sw, 1024, 256);
  k0_repack<<<4,    256, 0, stream>>>(Wsl,   wsl_sw,   32,  32);
  k0_zero  <<<66,   256, 0, stream>>>(stok_acc, 16896);   // stok + snorm contiguous

  // stage 1: projections + slice softmax + pooled-token partial sums
  k1_proj<<<1024, 256, 0, stream>>>(fx, ln1_s, ln1_b, bfx, bx, bsl, temp,
                                    wfx_sw, wx_sw, wsl_sw, sw_g, stok_acc, snorm);

  // stage 2: 32x32 attention over slice tokens, per (b,h)
  k2_attn<<<16, 256, 0, stream>>>(stok_acc, snorm, Wq, Wk, Wv, osl_sw);

  // stage 3: broadcast + out-proj + residual + LN2 + MLP + residual
  k3_out<<<1024, 256, 0, stream>>>(fx, sw_g, osl_sw, wout_sw, bout,
                                   ln2_s, ln2_b, wpre_sw, bpre, wpost_sw, bpost,
                                   (float*)d_out);
}
